// MessagePassingLayer_66194035965974
// MI455X (gfx1250) — matrix operand order repaired and checked by reading, not compiled, not run on hardware
//
#include <hip/hip_runtime.h>
#include <hip/hip_bf16.h>

typedef __attribute__((ext_vector_type(16))) _Float16 v16h;
typedef __attribute__((ext_vector_type(8)))  float    v8f;

constexpr int B_  = 2;
constexpr int N_  = 10000;
constexpr int E_  = 160000;
constexpr int ND  = 128;            // NODE_DIM
constexpr int ED  = 64;             // EDGE_DIM
constexpr int HID = 256;            // HIDDEN
constexpr int MSG_IN = 2 * ND + ED; // 320
constexpr int UPD_IN = ND + HID;    // 384
constexpr int TM  = 64;             // rows (edges/nodes) per workgroup tile

constexpr int LDA_E = MSG_IN + 8;   // 328 halves; bank-conflict-free stride
constexpr int LDA_N = UPD_IN + 8;   // 392 halves
constexpr int LDH_  = HID + 8;      // 264 halves

__device__ __forceinline__ float gelu_f(float x) {
  const float c = 0.7978845608028654f; // sqrt(2/pi)
  return 0.5f * x * (1.0f + tanhf(c * (x + 0.044715f * x * x * x)));
}

// B fragment: 16 contiguous halves; lane holds col = 16*tile + (lane&15),
// K = k0 + e with k0 including (lane>>4)*16 (contiguous k order, same as the A fragment).
__device__ __forceinline__ v16h load_b_frag(const _Float16* __restrict__ Wt,
                                            int ldk, int col, int k0) {
  return *reinterpret_cast<const v16h*>(Wt + (size_t)col * ldk + k0);
}

// A fragment from LDS: elements 0..15 at K = kt32 + hk*16 + e (contiguous).
__device__ __forceinline__ v16h load_a_frag(const _Float16* __restrict__ s,
                                            int stride, int row, int kt32, int hk) {
  const _Float16* p = s + row * stride + kt32 + hk * 16;
  v16h a;
  reinterpret_cast<uint4*>(&a)[0] = *reinterpret_cast<const uint4*>(p);
  reinterpret_cast<uint4*>(&a)[1] = *reinterpret_cast<const uint4*>(p + 8);
  return a;
}

__global__ void k_cvt_f16(const float* __restrict__ x, _Float16* __restrict__ y, size_t n) {
  size_t i = (size_t)blockIdx.x * blockDim.x + threadIdx.x;
  size_t stride = (size_t)gridDim.x * blockDim.x;
  for (; i < n; i += stride) y[i] = (_Float16)x[i];
}

// W[K][O] fp32 -> Wt[O][K] f16
__global__ void k_cvt_t(const float* __restrict__ W, _Float16* __restrict__ Wt, int K, int O) {
  int i = blockIdx.x * 256 + threadIdx.x;
  if (i < K * O) {
    int o = i / K, k = i - o * K;
    Wt[i] = (_Float16)W[(size_t)k * O + o];
  }
}

// ---------------- Edge message kernel ----------------
__global__ __launch_bounds__(512)
void k_edge(const _Float16* __restrict__ nodesH,
            const _Float16* __restrict__ edgesH,
            const int* __restrict__ eidx,
            const _Float16* __restrict__ W1T, const float* __restrict__ b1,
            const _Float16* __restrict__ W2T, const float* __restrict__ b2,
            float* __restrict__ agg) {
  __shared__ __align__(16) _Float16 sm[TM * LDA_E]; // 41984 B (reused for H tile)
  __shared__ int sSrc[TM], sDst[TM];
  const int tid = threadIdx.x;
  const int b   = blockIdx.y;
  const int e0  = blockIdx.x * TM;

  if (tid < TM) {
    const int2 p = reinterpret_cast<const int2*>(eidx)[(size_t)b * E_ + e0 + tid];
    sSrc[tid] = p.x; sDst[tid] = p.y;
  }
  __syncthreads();

  // Stage msg_in = [src(128) | dst(128) | ef(64)] as f16 rows of stride LDA_E.
  for (int i = tid; i < TM * 16; i += 512) {
    int e = i >> 4, c = i & 15;
    const uint4* ps = reinterpret_cast<const uint4*>(nodesH + ((size_t)b * N_ + sSrc[e]) * ND) + c;
    *reinterpret_cast<uint4*>(&sm[e * LDA_E + c * 8]) = *ps;
    const uint4* pd = reinterpret_cast<const uint4*>(nodesH + ((size_t)b * N_ + sDst[e]) * ND) + c;
    *reinterpret_cast<uint4*>(&sm[e * LDA_E + ND + c * 8]) = *pd;
  }
  {
    int e = tid >> 3, c = tid & 7; // 64 edges x 8 chunks == 512
    const uint4* pe = reinterpret_cast<const uint4*>(edgesH + ((size_t)b * E_ + e0 + e) * ED) + c;
    *reinterpret_cast<uint4*>(&sm[e * LDA_E + 2 * ND + c * 8]) = *pe;
  }
  __syncthreads();

  const int wv = tid >> 5, ln = tid & 31;
  const int col = wv * 16 + (ln & 15);
  const int hk  = ln >> 4;
  const int mr  = ln & 15;

  // GEMM1: [64 x 320] x [320 x 256]
  v8f acc[4] = {};
  for (int kt = 0; kt < MSG_IN / 32; ++kt) {
    v16h bf = load_b_frag(W1T, MSG_IN, col, kt * 32 + hk * 16);
    v16h af[4];
#pragma unroll
    for (int m = 0; m < 4; ++m)
      af[m] = load_a_frag(sm, LDA_E, m * 16 + mr, kt * 32, hk);
#pragma unroll
    for (int m = 0; m < 4; ++m)
      acc[m] = __builtin_amdgcn_wmma_f32_16x16x32_f16(false, af[m], false, bf,
                                                      (short)0, acc[m], false, false);
  }
  const float bias1 = b1[col];
  __syncthreads(); // everyone done reading A tile
#pragma unroll
  for (int m = 0; m < 4; ++m)
#pragma unroll
    for (int v = 0; v < 8; ++v) {
      int row = m * 16 + hk * 8 + v;
      sm[row * LDH_ + col] = (_Float16)gelu_f(acc[m][v] + bias1);
    }
  __syncthreads();

  // GEMM2: [64 x 256] x [256 x 256]
  v8f acc2[4] = {};
  for (int kt = 0; kt < HID / 32; ++kt) {
    v16h bf = load_b_frag(W2T, HID, col, kt * 32 + hk * 16);
    v16h af[4];
#pragma unroll
    for (int m = 0; m < 4; ++m)
      af[m] = load_a_frag(sm, LDH_, m * 16 + mr, kt * 32, hk);
#pragma unroll
    for (int m = 0; m < 4; ++m)
      acc2[m] = __builtin_amdgcn_wmma_f32_16x16x32_f16(false, af[m], false, bf,
                                                       (short)0, acc2[m], false, false);
  }
  const float bias2 = b2[col];
#pragma unroll
  for (int m = 0; m < 4; ++m)
#pragma unroll
    for (int v = 0; v < 8; ++v) {
      int row = m * 16 + hk * 8 + v;
      int d = sDst[row];
      unsafeAtomicAdd(&agg[((size_t)b * N_ + d) * HID + col], acc2[m][v] + bias2);
    }
}

// ---------------- Node update kernel ----------------
__global__ __launch_bounds__(512)
void k_node(const _Float16* __restrict__ nodesH,
            const float* __restrict__ agg,
            const _Float16* __restrict__ W1T, const float* __restrict__ b1,
            const _Float16* __restrict__ W2T, const float* __restrict__ b2,
            float* __restrict__ out) {
  __shared__ __align__(16) _Float16 sm[TM * LDA_N]; // 50176 B (reused for H tile)
  const int tid = threadIdx.x;
  const int b   = blockIdx.y;
  const int n0  = blockIdx.x * TM;

  // Stage upd_in = [node(128) | agg(256)->f16] rows of stride LDA_N.
  for (int i = tid; i < TM * 16; i += 512) {
    int r = i >> 4, c = i & 15;
    int node = min(n0 + r, N_ - 1);
    const uint4* p = reinterpret_cast<const uint4*>(nodesH + ((size_t)b * N_ + node) * ND) + c;
    *reinterpret_cast<uint4*>(&sm[r * LDA_N + c * 8]) = *p;
  }
  for (int i = tid; i < TM * 64; i += 512) {
    int r = i >> 6, c = i & 63;
    int node = min(n0 + r, N_ - 1);
    const float4 v = reinterpret_cast<const float4*>(agg + ((size_t)b * N_ + node) * HID)[c];
    _Float16* q = &sm[r * LDA_N + ND + c * 4];
    q[0] = (_Float16)v.x; q[1] = (_Float16)v.y; q[2] = (_Float16)v.z; q[3] = (_Float16)v.w;
  }
  __syncthreads();

  const int wv = tid >> 5, ln = tid & 31;
  const int hk = ln >> 4;
  const int mr = ln & 15;

  // GEMM1: [64 x 384] x [384 x 256]
  {
    const int col = wv * 16 + mr;
    v8f acc[4] = {};
    for (int kt = 0; kt < UPD_IN / 32; ++kt) {
      v16h bf = load_b_frag(W1T, UPD_IN, col, kt * 32 + hk * 16);
      v16h af[4];
#pragma unroll
      for (int m = 0; m < 4; ++m)
        af[m] = load_a_frag(sm, LDA_N, m * 16 + mr, kt * 32, hk);
#pragma unroll
      for (int m = 0; m < 4; ++m)
        acc[m] = __builtin_amdgcn_wmma_f32_16x16x32_f16(false, af[m], false, bf,
                                                        (short)0, acc[m], false, false);
    }
    const float bias1 = b1[col];
    __syncthreads();
#pragma unroll
    for (int m = 0; m < 4; ++m)
#pragma unroll
      for (int v = 0; v < 8; ++v) {
        int row = m * 16 + hk * 8 + v;
        sm[row * LDH_ + col] = (_Float16)gelu_f(acc[m][v] + bias1);
      }
    __syncthreads();
  }

  // GEMM2: [64 x 256] x [256 x 128] -> 8 col tiles x 2 row halves over 16 waves
  {
    const int col   = (wv & 7) * 16 + mr;
    const int mbase = (wv >> 3) * 2;
    v8f acc[2] = {};
    for (int kt = 0; kt < HID / 32; ++kt) {
      v16h bf = load_b_frag(W2T, HID, col, kt * 32 + hk * 16);
      v16h af[2];
#pragma unroll
      for (int m = 0; m < 2; ++m)
        af[m] = load_a_frag(sm, LDH_, (mbase + m) * 16 + mr, kt * 32, hk);
#pragma unroll
      for (int m = 0; m < 2; ++m)
        acc[m] = __builtin_amdgcn_wmma_f32_16x16x32_f16(false, af[m], false, bf,
                                                        (short)0, acc[m], false, false);
    }
    const float bias2 = b2[col];
#pragma unroll
    for (int m = 0; m < 2; ++m)
#pragma unroll
      for (int v = 0; v < 8; ++v) {
        int row  = (mbase + m) * 16 + hk * 8 + v;
        int node = n0 + row;
        if (node < N_)
          out[((size_t)b * N_ + node) * ND + col] = acc[m][v] + bias2;
      }
  }
}

extern "C" void kernel_launch(void* const* d_in, const int* in_sizes, int n_in,
                              void* d_out, int out_size, void* d_ws, size_t ws_size,
                              hipStream_t stream) {
  const float* nodeF = (const float*)d_in[0];
  const int*   eidx  = (const int*)d_in[1];
  const float* edgeF = (const float*)d_in[2];
  const float* W1m = (const float*)d_in[3];  const float* b1m = (const float*)d_in[4];
  const float* W2m = (const float*)d_in[5];  const float* b2m = (const float*)d_in[6];
  const float* W1u = (const float*)d_in[7];  const float* b1u = (const float*)d_in[8];
  const float* W2u = (const float*)d_in[9];  const float* b2u = (const float*)d_in[10];
  float* out = (float*)d_out;

  char* ws = (char*)d_ws;
  size_t off = 0;
  float*    agg    = (float*)(ws + off);    off += (size_t)B_ * N_ * HID * sizeof(float);
  _Float16* nodesH = (_Float16*)(ws + off); off += (size_t)B_ * N_ * ND * sizeof(_Float16);
  _Float16* edgesH = (_Float16*)(ws + off); off += (size_t)B_ * E_ * ED * sizeof(_Float16);
  _Float16* W1mT   = (_Float16*)(ws + off); off += (size_t)MSG_IN * HID * sizeof(_Float16);
  _Float16* W2mT   = (_Float16*)(ws + off); off += (size_t)HID * HID * sizeof(_Float16);
  _Float16* W1uT   = (_Float16*)(ws + off); off += (size_t)UPD_IN * HID * sizeof(_Float16);
  _Float16* W2uT   = (_Float16*)(ws + off); off += (size_t)HID * ND * sizeof(_Float16);

  // zero the scatter-add accumulator every launch (graph-replay safe)
  hipMemsetAsync(agg, 0, (size_t)B_ * N_ * HID * sizeof(float), stream);

  // f32 -> f16 feature conversions
  size_t nNode = (size_t)B_ * N_ * ND;
  size_t nEdge = (size_t)B_ * E_ * ED;
  k_cvt_f16<<<2048, 256, 0, stream>>>(nodeF, nodesH, nNode);
  k_cvt_f16<<<4096, 256, 0, stream>>>(edgeF, edgesH, nEdge);

  // weight transpose + convert: W[K][O] -> Wt[O][K]
  k_cvt_t<<<(MSG_IN * HID + 255) / 256, 256, 0, stream>>>(W1m, W1mT, MSG_IN, HID);
  k_cvt_t<<<(HID * HID + 255) / 256, 256, 0, stream>>>(W2m, W2mT, HID, HID);
  k_cvt_t<<<(UPD_IN * HID + 255) / 256, 256, 0, stream>>>(W1u, W1uT, UPD_IN, HID);
  k_cvt_t<<<(HID * ND + 255) / 256, 256, 0, stream>>>(W2u, W2uT, HID, ND);

  // edge message + scatter-add
  k_edge<<<dim3(E_ / TM, B_), 512, 0, stream>>>(nodesH, edgesH, eidx,
                                                W1mT, b1m, W2mT, b2m, agg);
  // node update
  k_node<<<dim3((N_ + TM - 1) / TM, B_), 512, 0, stream>>>(nodesH, agg,
                                                           W1uT, b1u, W2uT, b2u, out);
}
